// PointNetSetAbstractionMsg_446676598867
// MI455X (gfx1250) — matrix operand order repaired and checked by reading, not compiled, not run on hardware
//
#include <hip/hip_runtime.h>

// ---------------------------------------------------------------------------
// Problem constants (match reference)
// ---------------------------------------------------------------------------
#define BB 4
#define NN 8192
#define SS 1024
#define DD 6

typedef __attribute__((ext_vector_type(16))) _Float16 v16h;
typedef __attribute__((ext_vector_type(8)))  float    v8f;

// ---------------------------------------------------------------------------
// Workspace layout (bytes)
//   0      : maxt2   float[3][B]              (48 B)
//   64     : biases  float[672]               (2688 B)
//   2816   : weights _Float16[38912] (padded) (77824 B)
//   80896  : idx     int[458752]              (~1.75 MB)
// ---------------------------------------------------------------------------
#define WS_MAXT_OFF  0
#define WS_BIAS_OFF  64
#define WS_W_OFF     2816
#define WS_IDX_OFF   80896

// ---------------------------------------------------------------------------
// Kernel: per (branch, batch) time threshold  max_t^2
//   d[j] = t[j+1]-t[j];  tt[j] = d[j]+d[j+1] = t[j+2]-t[j]
//   sort tt descending, pick element 2*branch, square it.
// ---------------------------------------------------------------------------
__global__ void maxt_kernel(const float* __restrict__ t_list,
                            float* __restrict__ maxt2) {
  const int tid = threadIdx.x;
  if (tid >= 3 * BB) return;
  const int br = tid >> 2;
  const int b  = tid & 3;
  float t[7];
#pragma unroll
  for (int j = 0; j < 7; ++j) t[j] = t_list[b * 7 + j];
  float tt[5];
#pragma unroll
  for (int j = 0; j < 5; ++j) tt[j] = t[j + 2] - t[j];
#pragma unroll
  for (int a = 0; a < 5; ++a)
#pragma unroll
    for (int c = a + 1; c < 5; ++c)
      if (tt[c] > tt[a]) { float tmp = tt[a]; tt[a] = tt[c]; tt[c] = tmp; }
  const float v = tt[2 * br];
  maxt2[br * BB + b] = v * v;
}

// ---------------------------------------------------------------------------
// Kernel: fold batchnorm into weights (f16, C_in padded to mult of 32) + bias
//   y = relu(scale*(Wx+b) + (beta - mean*scale)),  scale = gamma/sqrt(var+eps)
//   W' = scale*W (f16),  b' = (b-mean)*scale + beta (f32)
// ---------------------------------------------------------------------------
__global__ void fold_kernel(const float* __restrict__ W, const float* __restrict__ b,
                            const float* __restrict__ beta, const float* __restrict__ gamma,
                            const float* __restrict__ mean, const float* __restrict__ var,
                            _Float16* __restrict__ Wdst, float* __restrict__ bdst,
                            int cout, int cin, int cinpad) {
  const int tid = blockIdx.x * blockDim.x + threadIdx.x;
  const int total = cout * cinpad;
  if (tid < total) {
    const int o = tid / cinpad;
    const int c = tid % cinpad;
    const float scale = gamma[o] * rsqrtf(var[o] + 1e-5f);
    Wdst[tid] = (c < cin) ? (_Float16)(W[o * cin + c] * scale) : (_Float16)0.0f;
  }
  if (tid < cout) {
    const float scale = gamma[tid] * rsqrtf(var[tid] + 1e-5f);
    bdst[tid] = (b[tid] - mean[tid]) * scale + beta[tid];
  }
}

// ---------------------------------------------------------------------------
// Kernel: ball query. One wave32 per (b,s) query. Lanes scan ascending index
// chunks; ballot+prefix compaction appends passing indices in ascending order
// (== reference's where/sort/take-first-K). Wave argmin of full 3D distance
// (first-occurrence tie-break) fills the remainder.
// ---------------------------------------------------------------------------
__global__ __launch_bounds__(128)
void query_kernel(const float* __restrict__ xyz, const float* __restrict__ mask,
                  const float* __restrict__ maxt2b, int* __restrict__ idxout,
                  float radius, int K) {
  const int wave = threadIdx.x >> 5;
  const int lane = threadIdx.x & 31;
  const int gid  = blockIdx.x * 4 + wave;        // gid = b*S + s
  const int b    = gid >> 10;
  const int s    = gid & (SS - 1);

  const float q0 = mask[(b * SS + s) * 3 + 0];
  const float q1 = mask[(b * SS + s) * 3 + 1];
  const float q2 = mask[(b * SS + s) * 3 + 2];
  const float mt2 = maxt2b[b];

  int   cnt  = 0;
  float bmin = 3.4e38f;
  int   bidx = 0;
  int* outp = idxout + (size_t)gid * K;

  const float* xb = xyz + (size_t)b * NN * 3;
  for (int it = 0; it < NN / 32; ++it) {
    const int j = it * 32 + lane;
    const float* px = xb + (size_t)j * 3;
    const float dx = px[0] - q0;
    const float dy = px[1] - q1;
    const float dz = px[2] - q2;
    const float dxy = dx * dx + dy * dy;   // squared xy distance
    const float dt  = dz * dz;             // squared t distance
    const float d3  = dxy + dt;            // squared 3D distance
    const bool good = (dxy <= radius) && (dt <= mt2);  // ref: bad = d>thr
    const unsigned mb = __builtin_amdgcn_ballot_w32(good);
    if (good) {
      const int pos = cnt + __popc(mb & ((1u << lane) - 1u));
      if (pos < K) outp[pos] = j;
    }
    cnt += __popc(mb);
    if (d3 < bmin) { bmin = d3; bidx = j; }  // strict <: keeps first occurrence
  }
  // wave argmin with first-index tie-break
#pragma unroll
  for (int off = 16; off; off >>= 1) {
    const float od = __shfl_xor(bmin, off, 32);
    const int   oi = __shfl_xor(bidx, off, 32);
    if (od < bmin || (od == bmin && oi < bidx)) { bmin = od; bidx = oi; }
  }
  for (int p = cnt + lane; p < K; p += 32) outp[p] = bidx;
}

// ---------------------------------------------------------------------------
// Kernel: new_xyz = transpose(mask) : out[b][c][s] = mask[b][s][c]
// ---------------------------------------------------------------------------
__global__ void xyz_transpose(const float* __restrict__ mask, float* __restrict__ out) {
  const int t = blockIdx.x * blockDim.x + threadIdx.x;
  if (t >= BB * 3 * SS) return;
  const int b = t / (3 * SS);
  const int r = t % (3 * SS);
  const int c = r / SS;
  const int s = r % SS;
  out[t] = mask[(b * SS + s) * 3 + c];
}

// ---------------------------------------------------------------------------
// One MLP layer on a [K_SAMP x CIN] f16 tile in LDS against folded f16 weights
// W[COUT][CIN] (CIN mult of 32). D = relu(A*W^T + bias). Non-final: store f16
// to Bout[K_SAMP][COUT] in LDS. Final: max over rows -> out channels.
//
// WMMA fragment layouts (CDNA5 ISA 7.12.2):
//   A 16x32 f16: lane L holds row M=(L&15); halves = K {kb..kb+15} (this code's order, not the ISA table's),
//                kb = (L<16 ? 0 : 16)
//   B 32x16 f16: lane L holds col N=(L&15); halves = K {kc..kc+15} (same order as A),
//                kc = (L<16 ? 0 : 16)
//   C/D 16x16 f32: lane L col N=(L&15); VGPR r -> row M = r + (L<16?0:8)
// ---------------------------------------------------------------------------
template<int K_SAMP, int CIN, int COUT, bool FINAL>
__device__ __forceinline__ void mlp_layer(const _Float16* __restrict__ A,
                                          _Float16* __restrict__ Bout,
                                          const _Float16* __restrict__ W,
                                          const float* __restrict__ bias,
                                          int lane,
                                          float* __restrict__ outp, int out_stride) {
  const int col = lane & 15;
  const int hi  = (lane & 16) ? 1 : 0;
#pragma unroll
  for (int nt = 0; nt < COUT / 16; ++nt) {
    const int o = nt * 16 + col;
    const float bs = bias[o];
    const _Float16* wrow = W + (size_t)o * CIN + hi * 16;
    v8f mx = {};
#pragma unroll
    for (int mt = 0; mt < K_SAMP / 16; ++mt) {
      v8f acc = {};
      const _Float16* arow = A + (mt * 16 + col) * CIN + hi * 16;
#pragma unroll
      for (int ks = 0; ks < CIN / 32; ++ks) {
        v16h a;
        const _Float16* ap = arow + ks * 32;
#pragma unroll
        for (int j = 0; j < 8; ++j) { a[j] = ap[j]; a[8 + j] = ap[8 + j]; }
        const v16h bf = *(const v16h*)(wrow + ks * 32);
        acc = __builtin_amdgcn_wmma_f32_16x16x32_f16(
            /*neg_a=*/false, a, /*neg_b=*/false, bf,
            /*c_mod=*/(short)0, acc, /*reuse_a=*/false, /*reuse_b=*/false);
      }
#pragma unroll
      for (int r = 0; r < 8; ++r) {
        float v = acc[r] + bs;
        v = v > 0.0f ? v : 0.0f;               // ReLU
        if (FINAL) {
          mx[r] = v > mx[r] ? v : mx[r];       // relu output >= 0, init 0 ok
        } else {
          const int row = mt * 16 + hi * 8 + r;
          Bout[row * COUT + o] = (_Float16)v;
        }
      }
    }
    if (FINAL) {
      float m = mx[0];
#pragma unroll
      for (int r = 1; r < 8; ++r) m = mx[r] > m ? mx[r] : m;
      const float om = __shfl_xor(m, 16, 32);  // combine rows 0-7 / 8-15 halves
      m = om > m ? om : m;
      if (lane < 16) outp[(size_t)o * out_stride] = m;
    }
  }
}

// ---------------------------------------------------------------------------
// Kernel: gather + 3-layer MLP + maxpool, one wave32 per (b,s) group.
// Block = 2 waves; per-wave ping-pong LDS buffers of K_SAMP x CMAX halves.
// ---------------------------------------------------------------------------
template<int K_SAMP, int C1, int C2, int C3, int CMAX>
__global__ __launch_bounds__(64)
void mlp_kernel(const float* __restrict__ xyz, const float* __restrict__ points,
                const float* __restrict__ mask, const int* __restrict__ idxbuf,
                const _Float16* __restrict__ W1, const _Float16* __restrict__ W2,
                const _Float16* __restrict__ W3,
                const float* __restrict__ b1, const float* __restrict__ b2,
                const float* __restrict__ b3,
                float* __restrict__ out_pts, int chbase) {
  __shared__ _Float16 smem[2 * 2 * K_SAMP * CMAX];
  const int wave = threadIdx.x >> 5;
  const int lane = threadIdx.x & 31;
  const int gid  = blockIdx.x * 2 + wave;       // gid = b*S + s
  const int b    = gid >> 10;
  const int s    = gid & (SS - 1);

  _Float16* bufA = smem + wave * (2 * K_SAMP * CMAX);
  _Float16* bufB = bufA + K_SAMP * CMAX;

  const int* gidx = idxbuf + (size_t)gid * K_SAMP;
  const float q0 = mask[(b * SS + s) * 3 + 0];
  const float q1 = mask[(b * SS + s) * 3 + 1];
  const float q2 = mask[(b * SS + s) * 3 + 2];

  // Gather: [K][32] with channels 0..5 = points, 6..8 = xyz - query, 9..31 = 0
  for (int k = lane; k < K_SAMP; k += 32) {
    const int i = gidx[k];
    const float* pp = points + ((size_t)b * NN + i) * DD;
    const float* px = xyz    + ((size_t)b * NN + i) * 3;
    _Float16* row = bufA + k * 32;
#pragma unroll
    for (int c = 0; c < DD; ++c) row[c] = (_Float16)pp[c];
    row[6] = (_Float16)(px[0] - q0);
    row[7] = (_Float16)(px[1] - q1);
    row[8] = (_Float16)(px[2] - q2);
#pragma unroll
    for (int c = 9; c < 32; ++c) row[c] = (_Float16)0.0f;
  }
  __syncthreads();

  mlp_layer<K_SAMP, 32, C1, false>(bufA, bufB, W1, b1, lane, nullptr, 0);
  __syncthreads();
  mlp_layer<K_SAMP, C1, C2, false>(bufB, bufA, W2, b2, lane, nullptr, 0);
  __syncthreads();

  float* outp = out_pts + ((size_t)b * 320 + chbase) * SS + s;
  mlp_layer<K_SAMP, C2, C3, true>(bufA, nullptr, W3, b3, lane, outp, SS);
}

// ---------------------------------------------------------------------------
// Host launcher
// ---------------------------------------------------------------------------
extern "C" void kernel_launch(void* const* d_in, const int* in_sizes, int n_in,
                              void* d_out, int out_size, void* d_ws, size_t ws_size,
                              hipStream_t stream) {
  (void)in_sizes; (void)n_in; (void)out_size; (void)ws_size;

  const float* xyz    = (const float*)d_in[0];   // [B,N,3]
  const float* points = (const float*)d_in[1];   // [B,N,D]
  const float* mask   = (const float*)d_in[2];   // [B,S,3]
  const float* t_list = (const float*)d_in[3];   // [B,7]
  // params pytree flattened: per (branch,layer), dict keys sorted:
  //   'W','b','beta','gamma','mean','var'

  char* ws = (char*)d_ws;
  float*    maxt2  = (float*)(ws + WS_MAXT_OFF);   // [3][B]
  float*    biases = (float*)(ws + WS_BIAS_OFF);   // 672 floats
  _Float16* wbuf   = (_Float16*)(ws + WS_W_OFF);   // 38912 halves (padded)
  int*      idxbuf = (int*)(ws + WS_IDX_OFF);      // 458752 ints

  static const int CIN[9]  = {9, 32, 32,   9, 64, 64,   9, 64, 96};
  static const int CINP[9] = {32, 32, 32,  32, 64, 64,  32, 64, 96};
  static const int COUT[9] = {32, 32, 64,  64, 64, 128, 64, 96, 128};
  static const int WOFF[9] = {0, 1024, 2048, 4096, 6144, 10240, 18432, 20480, 26624};
  static const int BOFF[9] = {0, 32, 64, 128, 192, 256, 384, 448, 544};

  maxt_kernel<<<1, 32, 0, stream>>>(t_list, maxt2);

  for (int l = 0; l < 9; ++l) {
    const float* W     = (const float*)d_in[4 + l * 6 + 0];
    const float* bvec  = (const float*)d_in[4 + l * 6 + 1];
    const float* beta  = (const float*)d_in[4 + l * 6 + 2];
    const float* gamma = (const float*)d_in[4 + l * 6 + 3];
    const float* mean  = (const float*)d_in[4 + l * 6 + 4];
    const float* var   = (const float*)d_in[4 + l * 6 + 5];
    const int total = COUT[l] * CINP[l];
    fold_kernel<<<(total + 255) / 256, 256, 0, stream>>>(
        W, bvec, beta, gamma, mean, var,
        wbuf + WOFF[l], biases + BOFF[l], COUT[l], CIN[l], CINP[l]);
  }

  static const float radii[3] = {0.1f, 0.2f, 0.4f};
  static const int   KK[3]    = {16, 32, 64};
  static const size_t IOFF[3] = {0, 65536, 196608};
  for (int br = 0; br < 3; ++br) {
    query_kernel<<<(BB * SS) / 4, 128, 0, stream>>>(
        xyz, mask, maxt2 + br * BB, idxbuf + IOFF[br], radii[br], KK[br]);
  }

  float* out = (float*)d_out;
  xyz_transpose<<<(BB * 3 * SS + 255) / 256, 256, 0, stream>>>(mask, out);
  float* out_pts = out + BB * 3 * SS;   // new_points [B,320,S]

  mlp_kernel<16, 32, 32, 64, 32><<<(BB * SS) / 2, 64, 0, stream>>>(
      xyz, points, mask, idxbuf + IOFF[0],
      wbuf + WOFF[0], wbuf + WOFF[1], wbuf + WOFF[2],
      biases + BOFF[0], biases + BOFF[1], biases + BOFF[2], out_pts, 0);

  mlp_kernel<32, 64, 64, 128, 64><<<(BB * SS) / 2, 64, 0, stream>>>(
      xyz, points, mask, idxbuf + IOFF[1],
      wbuf + WOFF[3], wbuf + WOFF[4], wbuf + WOFF[5],
      biases + BOFF[3], biases + BOFF[4], biases + BOFF[5], out_pts, 64);

  mlp_kernel<64, 64, 96, 128, 96><<<(BB * SS) / 2, 64, 0, stream>>>(
      xyz, points, mask, idxbuf + IOFF[2],
      wbuf + WOFF[6], wbuf + WOFF[7], wbuf + WOFF[8],
      biases + BOFF[6], biases + BOFF[7], biases + BOFF[8], out_pts, 192);
}
